// Encoder_Decoder_30580167147776
// MI455X (gfx1250) — matrix operand order repaired and checked by reading, not compiled, not run on hardware
//
#include <hip/hip_runtime.h>
#include <hip/hip_bf16.h>
#include <math.h>

// ---------------------------------------------------------------------------
// Types for CDNA5 WMMA (wave32): v_wmma_f32_16x16x32_bf16
// ---------------------------------------------------------------------------
typedef __attribute__((ext_vector_type(16))) __bf16 v16bf;
typedef __attribute__((ext_vector_type(8)))  __bf16 v8bf;
typedef __attribute__((ext_vector_type(2)))  __bf16 v2bf;
typedef __attribute__((ext_vector_type(8)))  float  v8f;

#define HDIM 128
#define NCLS_ 80

__device__ __forceinline__ float sigf(float x) { return 1.0f / (1.0f + __expf(-x)); }

// pack two f32 -> one dword of 2 bf16 (v_cvt_pk_bf16_f32)
__device__ __forceinline__ unsigned pk2(float a, float b) {
    v2bf p; p[0] = (__bf16)a; p[1] = (__bf16)b;
    return __builtin_bit_cast(unsigned, p);
}

// ---------------------------------------------------------------------------
// Fused GEMM: C[M,N] = act(A[M,K] @ W[N,K]^T + bias[N])
// Block = 256 threads (8 waves). Block tile = 64 rows x 64 cols, K chunks of 32.
// Wave (wr = wave&3, wc = wave>>2) computes rows [wr*16..+16) x cols [wc*32..+32)
// => 2 WMMAs per wave per K chunk. Staging is branch-free float4 loads with
// address clamping + value select, packed bf16 dword stores to LDS, plus
// global_prefetch of the next K chunk.
// Requirements (met by all call sites): K % 32 == 0, N % 64 == 0, bias != null.
// ---------------------------------------------------------------------------
__global__ __launch_bounds__(256)
void gemm_bias_act(const float* __restrict__ A, int lda,
                   const float* __restrict__ W,      // [N,K] row-major
                   const float* __restrict__ bias,   // [N]
                   float* __restrict__ C, int ldc,
                   int Mrows, int K, int act)
{
    __shared__ __align__(16) unsigned At[64][16];   // 64 rows x 32 bf16
    __shared__ __align__(16) unsigned Wt[64][16];   // 64 cols x 32 bf16

    const int tid  = threadIdx.x;
    const int lane = tid & 31;
    const int wave = tid >> 5;
    const int half = lane >> 4;
    const int l16  = lane & 15;
    const int wr   = wave & 3;
    const int wc   = wave >> 2;
    const int row0 = blockIdx.x * 64;
    const int col0 = blockIdx.y * 64;

    const int sr  = tid >> 3;          // 0..31 (two passes cover 64 rows)
    const int sk4 = tid & 7;           // float4 index within the 32-wide chunk

    v8f acc0 = {}, acc1 = {};

    for (int k0 = 0; k0 < K; k0 += 32) {
        #pragma unroll
        for (int i = 0; i < 2; ++i) {
            const int r  = sr + i * 32;
            const int gr = row0 + r;
            const int cr = (gr < Mrows) ? gr : (Mrows - 1);
            const float* src = &A[(size_t)cr * lda + k0 + sk4 * 4];
            float4 v = *(const float4*)src;
            if (k0 + 32 < K) __builtin_prefetch(src + 32, 0, 0);
            const bool ok = gr < Mrows;
            v.x = ok ? v.x : 0.0f; v.y = ok ? v.y : 0.0f;
            v.z = ok ? v.z : 0.0f; v.w = ok ? v.w : 0.0f;
            uint2 p; p.x = pk2(v.x, v.y); p.y = pk2(v.z, v.w);
            *(uint2*)&At[r][sk4 * 2] = p;
        }
        #pragma unroll
        for (int i = 0; i < 2; ++i) {
            const int r = sr + i * 32;                // column index within tile
            const float* src = &W[(size_t)(col0 + r) * K + k0 + sk4 * 4];
            float4 v = *(const float4*)src;
            if (k0 + 32 < K) __builtin_prefetch(src + 32, 0, 0);
            uint2 p; p.x = pk2(v.x, v.y); p.y = pk2(v.z, v.w);
            *(uint2*)&Wt[r][sk4 * 2] = p;
        }
        __syncthreads();

        // A fragment: row = wr*16 + l16; K pattern {kb..kb+7, kb+16..kb+23}
        const __bf16* Ar = (const __bf16*)&At[wr * 16 + l16][0];
        const int kb = half ? 8 : 0;
        v8bf alo = *(const v8bf*)(Ar + kb);
        v8bf ahi = *(const v8bf*)(Ar + kb + 16);
        v16bf afrag = __builtin_shufflevector(alo, ahi,
                        0,1,2,3,4,5,6,7,8,9,10,11,12,13,14,15);
        // B fragments: col = wc*32 + {0,16} + l16; K pattern {kb..kb+7, kb+16..kb+23}
        const __bf16* B0 = (const __bf16*)&Wt[wc * 32 + l16][0] + kb;
        const __bf16* B1 = (const __bf16*)&Wt[wc * 32 + 16 + l16][0] + kb;
        v16bf bf0 = __builtin_shufflevector(*(const v8bf*)B0, *(const v8bf*)(B0 + 16),
                        0,1,2,3,4,5,6,7,8,9,10,11,12,13,14,15);
        v16bf bf1 = __builtin_shufflevector(*(const v8bf*)B1, *(const v8bf*)(B1 + 16),
                        0,1,2,3,4,5,6,7,8,9,10,11,12,13,14,15);

        acc0 = __builtin_amdgcn_wmma_f32_16x16x32_bf16(false, afrag, false, bf0,
                                                       (short)0, acc0, false, false);
        acc1 = __builtin_amdgcn_wmma_f32_16x16x32_bf16(false, afrag, false, bf1,
                                                       (short)0, acc1, false, false);
        __syncthreads();
    }

    // epilogue: VGPR j -> M = j + half*8, N = l16
    const int n0 = col0 + wc * 32 + l16;
    const float b0 = bias[n0];
    const float b1 = bias[n0 + 16];
    #pragma unroll
    for (int j = 0; j < 8; ++j) {
        const int m = row0 + wr * 16 + j + half * 8;
        if (m < Mrows) {
            float v0 = acc0[j] + b0, v1 = acc1[j] + b1;
            if (act) { v0 = fmaxf(v0, 0.0f); v1 = fmaxf(v1, 0.0f); }
            C[(size_t)m * ldc + n0]      = v0;
            C[(size_t)m * ldc + n0 + 16] = v1;
        }
    }
}

// ---------------------------------------------------------------------------
// Row-wise 3-way concat in float4 units (all widths % 4 == 0)
// ---------------------------------------------------------------------------
__global__ void concat3(const float4* __restrict__ a, int wa4,
                        const float4* __restrict__ b, int wb4,
                        const float4* __restrict__ c, int wc4,
                        float4* __restrict__ out, int rows)
{
    const int w4 = wa4 + wb4 + wc4;
    const size_t total = (size_t)rows * w4;
    for (size_t i = (size_t)blockIdx.x * blockDim.x + threadIdx.x; i < total;
         i += (size_t)gridDim.x * blockDim.x) {
        const int r = (int)(i / w4), col = (int)(i % w4);
        float4 v;
        if (col < wa4)             v = a[(size_t)r * wa4 + col];
        else if (col < wa4 + wb4)  v = b[(size_t)r * wb4 + (col - wa4)];
        else                       v = c[(size_t)r * wc4 + (col - wa4 - wb4)];
        out[i] = v;
    }
}

// ---------------------------------------------------------------------------
// Sequence-tile descriptors + shared zero vector.
// Per block: 40 ints: [0]=nseq [1]=stride [2]=maxlen [3..18]=base[16] [19..34]=len[16]
// Blocks 0..1: encoder fwd/bwd. Blocks 2..11: decoder fwd x5, bwd x5.
// ---------------------------------------------------------------------------
__global__ void setup_desc(const int* __restrict__ ucl, int* __restrict__ desc,
                           float* __restrict__ zeros, int N)
{
    if (blockIdx.x != 0) return;
    if (threadIdx.x < HDIM) zeros[threadIdx.x] = 0.0f;
    if (threadIdx.x != 0) return;
    for (int d = 0; d < 2; ++d) {
        int* D = desc + d * 40;
        D[0] = 1; D[1] = d ? -1 : 1; D[2] = N;
        for (int s = 0; s < 16; ++s) { D[3 + s] = d ? (N - 1) : 0; D[19 + s] = (s == 0) ? N : 0; }
    }
    for (int d = 0; d < 2; ++d)
        for (int jt = 0; jt < 5; ++jt) {
            int* D = desc + (2 + d * 5 + jt) * 40;
            D[1] = d ? -1 : 1;
            int ml = 0, ns = 0;
            for (int q = 0; q < 16; ++q) {
                int i = jt * 16 + q;
                int base = 0, len = 0;
                if (i < NCLS_) {
                    int s0 = ucl[i], s1 = ucl[i + 1];
                    len = s1 - s0;
                    base = d ? (s1 - 1) : s0;
                    ns++;
                }
                D[3 + q] = base; D[19 + q] = len;
                if (len > ml) ml = len;
            }
            D[0] = ns; D[2] = ml;
        }
}

// ---------------------------------------------------------------------------
// Batched GRU scan: one block = up to 16 independent sequences = rows of the
// WMMA A tile. Per step: gates = h @ Whh^T + bhh with 12 WMMAs/wave (8 waves
// cover 384 gate cols; Whh lives in VGPRs as pre-built bf16 B-fragments),
// then gate nonlinearities + hidden update through LDS.
// ---------------------------------------------------------------------------
__global__ __launch_bounds__(256)
void gru_scan(const int* __restrict__ desc_all, int tilesPerDir,
              const float* __restrict__ Whh0, const float* __restrict__ Whh1,
              const float* __restrict__ bhh0, const float* __restrict__ bhh1,
              const float* __restrict__ h00,  const float* __restrict__ h01,
              const float* __restrict__ gx0,  const float* __restrict__ gx1,
              float* __restrict__ ys, int ysW, int colOff0, int colOff1,
              float* __restrict__ hT0, float* __restrict__ hT1)
{
    __shared__ float gates[16][3 * HDIM];             // 24 KB
    __shared__ float hcur[16][HDIM];                  //  8 KB
    __shared__ __align__(32) __bf16 hbf[16][HDIM];    //  4 KB
    __shared__ int sbase[16], slen[16];

    const int dir = (blockIdx.x >= tilesPerDir) ? 1 : 0;
    const int* D        = desc_all + blockIdx.x * 40;
    const float* Whh    = dir ? Whh1 : Whh0;
    const float* bhh    = dir ? bhh1 : bhh0;
    const float* h0     = dir ? h01  : h00;
    const float* gx     = dir ? gx1  : gx0;
    const int    colOff = dir ? colOff1 : colOff0;
    float*       hT     = dir ? hT1  : hT0;

    const int tid  = threadIdx.x;
    const int lane = tid & 31, wave = tid >> 5;
    const int half = lane >> 4, l16 = lane & 15;
    const int stride = D[1];
    const int maxT   = D[2];

    if (tid < 16) { sbase[tid] = D[3 + tid]; slen[tid] = D[19 + tid]; }
    for (int p = tid; p < 16 * HDIM; p += 256) {
        int s = p >> 7, j = p & (HDIM - 1);
        float v = h0[j];
        hcur[s][j] = v; hbf[s][j] = (__bf16)v;
    }

    // Preload Whh^T as B fragments (wave covers gate cols [wave*48, wave*48+48))
    v16bf bw[3][4];
    float bias[3];
    #pragma unroll
    for (int t = 0; t < 3; ++t) {
        const int n = wave * 48 + t * 16 + l16;
        bias[t] = bhh[n];
        #pragma unroll
        for (int c = 0; c < 4; ++c) {
            const int kb = c * 32 + (half ? 16 : 0);
            v16bf f;
            #pragma unroll
            for (int i = 0; i < 16; ++i)
                f[i] = (__bf16)Whh[(size_t)n * HDIM + (kb + i)];
            bw[t][c] = f;
        }
    }
    __syncthreads();

    for (int step = 0; step < maxT; ++step) {
        // ---- recurrent matmul: gates = hcur @ Whh^T + bhh ----
        v8f acc[3];
        #pragma unroll
        for (int t = 0; t < 3; ++t) {
            v8f a;
            #pragma unroll
            for (int j = 0; j < 8; ++j) a[j] = bias[t];
            acc[t] = a;
        }
        #pragma unroll
        for (int c = 0; c < 4; ++c) {
            const int kb = c * 32 + (half ? 16 : 0);
            v8bf lo = *(const v8bf*)&hbf[l16][kb];
            v8bf hi = *(const v8bf*)&hbf[l16][kb + 8];
            v16bf af = __builtin_shufflevector(lo, hi,
                         0,1,2,3,4,5,6,7,8,9,10,11,12,13,14,15);
            #pragma unroll
            for (int t = 0; t < 3; ++t)
                acc[t] = __builtin_amdgcn_wmma_f32_16x16x32_bf16(
                             false, af, false, bw[t][c], (short)0, acc[t], false, false);
        }
        #pragma unroll
        for (int t = 0; t < 3; ++t) {
            const int n = wave * 48 + t * 16 + l16;
            #pragma unroll
            for (int j = 0; j < 8; ++j)
                gates[j + half * 8][n] = acc[t][j];
        }
        __syncthreads();

        // ---- gate nonlinearities + hidden update ----
        for (int p = tid; p < 16 * HDIM; p += 256) {
            const int s = p >> 7, j = p & (HDIM - 1);
            const int len = slen[s];
            const bool active = step < len;
            const int row = sbase[s] + (active ? step * stride : 0);
            const float* g = gx + (size_t)row * (3 * HDIM);
            const float rg = sigf(g[j]             + gates[s][j]);
            const float z  = sigf(g[HDIM + j]      + gates[s][HDIM + j]);
            const float nn = tanhf(g[2 * HDIM + j] + rg * gates[s][2 * HDIM + j]);
            const float h  = hcur[s][j];
            const float hnew = (1.0f - z) * nn + z * h;
            hcur[s][j] = hnew;
            hbf[s][j]  = (__bf16)hnew;
            if (active && ys) ys[(size_t)row * ysW + colOff + j] = hnew;
        }
        __syncthreads();
    }

    if (hT && tid < HDIM) hT[tid] = hcur[0][tid];
}

// ---------------------------------------------------------------------------
// out[m] = sigmoid(dec_out[m,:256] . out_W + out_b)   (float4 dot)
// ---------------------------------------------------------------------------
__global__ void out_sigmoid(const float4* __restrict__ dec_out,
                            const float4* __restrict__ outW,
                            const float* __restrict__ outb,
                            float* __restrict__ out, int Mrows)
{
    const int m = blockIdx.x * blockDim.x + threadIdx.x;
    if (m >= Mrows) return;
    float s = outb[0];
    const float4* r = dec_out + (size_t)m * 64;
    #pragma unroll 8
    for (int c = 0; c < 64; ++c) {
        float4 rv = r[c], wv = outW[c];
        s += rv.x * wv.x + rv.y * wv.y + rv.z * wv.z + rv.w * wv.w;
    }
    out[m] = sigf(s);
}

// ---------------------------------------------------------------------------
extern "C" void kernel_launch(void* const* d_in, const int* in_sizes, int n_in,
                              void* d_out, int out_size, void* d_ws, size_t ws_size,
                              hipStream_t stream)
{
    const float* boxes_feature = (const float*)d_in[0];
    const float* boxes_score   = (const float*)d_in[1];
    const float* boxes_box     = (const float*)d_in[2];
    const float* acb_feature   = (const float*)d_in[3];
    const float* acb_score     = (const float*)d_in[4];
    const float* acb_box       = (const float*)d_in[5];
    const int*   ucl           = (const int*)d_in[7];
    const float* appear_W = (const float*)d_in[8];
    const float* appear_b = (const float*)d_in[9];
    const float* s1_W = (const float*)d_in[10];
    const float* s1_b = (const float*)d_in[11];
    const float* s2_W = (const float*)d_in[12];
    const float* s2_b = (const float*)d_in[13];
    const float* box_W = (const float*)d_in[14];
    const float* box_b = (const float*)d_in[15];
    const float* encf_W = (const float*)d_in[16];
    const float* encf_b = (const float*)d_in[17];
    const float* decf_W = (const float*)d_in[18];
    const float* decf_b = (const float*)d_in[19];
    const float* out_W = (const float*)d_in[20];
    const float* out_b = (const float*)d_in[21];
    const float* enc_Wih = (const float*)d_in[22];
    const float* enc_Whh = (const float*)d_in[23];
    const float* enc_bih = (const float*)d_in[24];
    const float* enc_bhh = (const float*)d_in[25];
    const float* dec_Wih = (const float*)d_in[26];
    const float* dec_Whh = (const float*)d_in[27];
    const float* dec_bih = (const float*)d_in[28];
    const float* dec_bhh = (const float*)d_in[29];

    const int N = in_sizes[0] / 1024;
    const int M = in_sizes[3] / 1024;

    // ---- workspace layout (fp32) ----
    float* f = (float*)d_ws;
    size_t off = 0;
    auto alloc = [&](size_t n) { float* p = f + off; off += n; return p; };
    float* enc_feat  = alloc((size_t)N * 128);
    float* dec_feat  = alloc((size_t)M * 128);
    float* s1buf     = alloc((size_t)N * 512);   // reused as enc_cat after enc_score
    float* enc_score = alloc((size_t)N * 128);
    float* enc_box   = alloc((size_t)N * 128);
    float* enc_all   = alloc((size_t)N * 128);
    float* dec_cat   = alloc((size_t)M * 192);
    float* dec_all   = alloc((size_t)M * 128);
    float* gx_ef     = alloc((size_t)N * 384);   // reused as dec_out after enc scan
    float* gx_eb     = alloc((size_t)N * 384);
    float* gx_df     = alloc((size_t)M * 384);
    float* gx_db     = alloc((size_t)M * 384);
    float* hf        = alloc(128);
    float* hb        = alloc(128);
    float* zeros     = alloc(128);
    int*   desc      = (int*)(f + off); off += 512;
    float* enc_cat   = s1buf;                    // alias (s1 consumed by then)
    float* dec_out   = gx_ef;                    // alias (gx_ef consumed by then)

    setup_desc<<<1, 128, 0, stream>>>(ucl, desc, zeros, N);

    dim3 blk(256);
    auto gemm = [&](const float* A, int lda, const float* W, const float* b,
                    float* C, int ldc, int Mr, int Nc, int K, int act) {
        dim3 grid((Mr + 63) / 64, Nc / 64);
        gemm_bias_act<<<grid, blk, 0, stream>>>(A, lda, W, b, C, ldc, Mr, K, act);
    };

    // feed-forward stage
    gemm(boxes_feature, 1024, appear_W, appear_b, enc_feat, 128, N, 128, 1024, 1);
    gemm(acb_feature,   1024, appear_W, appear_b, dec_feat, 128, M, 128, 1024, 1);
    gemm(boxes_score,   2560, s1_W,     s1_b,     s1buf,    512, N, 512, 2560, 1);
    gemm(s1buf,          512, s2_W,     s2_b,     enc_score,128, N, 128,  512, 1);
    gemm(boxes_box,      320, box_W,    box_b,    enc_box,  128, N, 128,  320, 1);

    concat3<<<1024, 256, 0, stream>>>((const float4*)enc_feat, 32,
                                      (const float4*)enc_score, 32,
                                      (const float4*)enc_box, 32,
                                      (float4*)enc_cat, N);
    gemm(enc_cat, 384, encf_W, encf_b, enc_all, 128, N, 128, 384, 1);

    concat3<<<1024, 256, 0, stream>>>((const float4*)dec_feat, 32,
                                      (const float4*)acb_score, 8,
                                      (const float4*)acb_box, 8,
                                      (float4*)dec_cat, M);
    gemm(dec_cat, 192, decf_W, decf_b, dec_all, 128, M, 128, 192, 1);

    // input-side GRU gates (time-parallel)
    gemm(enc_all, 128, enc_Wih,             enc_bih,       gx_ef, 384, N, 384, 128, 0);
    gemm(enc_all, 128, enc_Wih + 384 * 128, enc_bih + 384, gx_eb, 384, N, 384, 128, 0);
    gemm(dec_all, 128, dec_Wih,             dec_bih,       gx_df, 384, M, 384, 128, 0);
    gemm(dec_all, 128, dec_Wih + 384 * 128, dec_bih + 384, gx_db, 384, M, 384, 128, 0);

    // encoder bi-GRU: 2 blocks (fwd, bwd) run concurrently
    gru_scan<<<2, 256, 0, stream>>>(desc, 1,
        enc_Whh, enc_Whh + 384 * 128, enc_bhh, enc_bhh + 384,
        zeros, zeros, gx_ef, gx_eb,
        nullptr, 0, 0, 0, hf, hb);

    // decoder bi-GRU: 80 independent segments/dir, 16 per tile, 10 blocks
    gru_scan<<<10, 256, 0, stream>>>(desc + 2 * 40, 5,
        dec_Whh, dec_Whh + 384 * 128, dec_bhh, dec_bhh + 384,
        hf, hb, gx_df, gx_db,
        dec_out, 256, 0, 128, nullptr, nullptr);

    out_sigmoid<<<(M + 255) / 256, 256, 0, stream>>>((const float4*)dec_out,
                                                     (const float4*)out_W, out_b,
                                                     (float*)d_out, M);
}
